// EvenLayerTransformerEncoder_58772332478837
// MI455X (gfx1250) — matrix operand order repaired and checked by reading, not compiled, not run on hardware
//
#include <hip/hip_runtime.h>
#include <math.h>

// ---------------------------------------------------------------------------
// CDNA5 WMMA types (gfx1250, wave32): D(f32,16x16) = A(bf16,16x32) x B(bf16,32x16) + C
// ---------------------------------------------------------------------------
typedef __attribute__((ext_vector_type(16))) __bf16 v16bf;
typedef __attribute__((ext_vector_type(8)))  __bf16 v8bf;
typedef __attribute__((ext_vector_type(8)))  float  v8f;

__device__ __forceinline__ v8f wmma_bf16(v16bf a, v16bf b, v8f c) {
  return __builtin_amdgcn_wmma_f32_16x16x32_bf16(false, a, false, b, (short)0, c,
                                                 false, false);
}

// Load a 16-element bf16 fragment from two contiguous 8-element (16B) runs.
__device__ __forceinline__ v16bf ld_frag(const __bf16* p0, const __bf16* p1) {
  v8bf lo = *(const v8bf*)p0;
  v8bf hi = *(const v8bf*)p1;
  return __builtin_shufflevector(lo, hi, 0, 1, 2, 3, 4, 5, 6, 7, 8, 9, 10, 11,
                                 12, 13, 14, 15);
}

__device__ __forceinline__ v8bf pack8(float4 a, float4 b) {
  v8bf r;
  r[0] = (__bf16)a.x; r[1] = (__bf16)a.y; r[2] = (__bf16)a.z; r[3] = (__bf16)a.w;
  r[4] = (__bf16)b.x; r[5] = (__bf16)b.y; r[6] = (__bf16)b.z; r[7] = (__bf16)b.w;
  return r;
}

__device__ __forceinline__ v16bf pack16(float4 a, float4 b, float4 c, float4 d) {
  return __builtin_shufflevector(pack8(a, b), pack8(c, d), 0, 1, 2, 3, 4, 5, 6,
                                 7, 8, 9, 10, 11, 12, 13, 14, 15);
}

// ---------------------------------------------------------------------------
// Generic GEMM: C[M,N] = act(scale * (A[M,K] @ W[K,N] + bias[N]))
// Requires K % 32 == 0 and N % 128 == 0 (true for all uses here).
// block = 256 threads (8 waves), tile 64(M) x 128(N), waves 4(M) x 2(N),
// each wave -> 16x64 of C via four 16x16 WMMA accumulators, K step 32.
// Double-buffered LDS (one barrier per K-step); W stored transposed so
// B-fragments are two contiguous 8-element k runs. All fragments are hoisted
// before the WMMA sequence so the 4 matrix ops issue back-to-back.
// ---------------------------------------------------------------------------
__global__ __launch_bounds__(256) void gemm_bias_kernel(
    const float* __restrict__ A, const float* __restrict__ W,
    const float* __restrict__ bias, float* __restrict__ C, int M, int N, int K,
    float scale, int relu) {
  __shared__ __bf16 As[2][64][40];    // 80B row stride: 16B aligned, non-pow2
  __shared__ __bf16 Wt[2][128][40];   // transposed W tile: Wt[n][k]

  const int tid  = threadIdx.x;
  const int lane = tid & 31;
  const int wave = tid >> 5;
  const int wm   = wave & 3;
  const int wn   = wave >> 2;
  const int m0   = blockIdx.y * 64;
  const int n0   = blockIdx.x * 128;
  const int row  = lane & 15;
  const int kg   = lane >> 4;

  // staging index precompute
  const int ar  = tid >> 2;            // A: row 0..63
  const int ac0 = (tid & 3) * 8;       // A: k offset 0/8/16/24
  const int agm = m0 + ar;
  const size_t arow_g = (size_t)(agm < M ? agm : (M - 1)) * K;
  const bool aok = agm < M;

  const int wcol = tid & 127;          // W: n 0..127
  const int wkq  = (tid >> 7) * 16;    // W: k offset 0 or 16

  // prefetch registers
  float4 pa0, pa1;
  float  pw[16];

  auto loadTile = [&](int k0) {
    const float* ap = A + arow_g + k0 + ac0;
    pa0 = *(const float4*)ap;
    pa1 = *(const float4*)(ap + 4);
    if (!aok) {
      pa0 = make_float4(0.f, 0.f, 0.f, 0.f);
      pa1 = pa0;
    }
    const float* wp = W + (size_t)(k0 + wkq) * N + n0 + wcol;
#pragma unroll
    for (int c = 0; c < 16; ++c) pw[c] = wp[(size_t)c * N];
  };
  auto storeTile = [&](int bufi) {
    *(v8bf*)&As[bufi][ar][ac0] = pack8(pa0, pa1);
    *(v8bf*)&Wt[bufi][wcol][wkq] =
        pack8(make_float4(pw[0], pw[1], pw[2], pw[3]),
              make_float4(pw[4], pw[5], pw[6], pw[7]));
    *(v8bf*)&Wt[bufi][wcol][wkq + 8] =
        pack8(make_float4(pw[8], pw[9], pw[10], pw[11]),
              make_float4(pw[12], pw[13], pw[14], pw[15]));
  };

  v8f acc[4] = {};

  loadTile(0);
  storeTile(0);
  int cur = 0;
  for (int k0 = 0; k0 < K; k0 += 32) {
    __syncthreads();
    const bool more = (k0 + 32) < K;
    if (more) loadTile(k0 + 32);  // overlap global latency with WMMAs

    // hoist all fragment loads, then issue 4 WMMAs back-to-back
    const int arow = wm * 16 + row;
    v16bf af = ld_frag(&As[cur][arow][kg * 8], &As[cur][arow][16 + kg * 8]);
    v16bf bf[4];
#pragma unroll
    for (int t = 0; t < 4; ++t) {
      const int cn = wn * 64 + t * 16 + row;
      bf[t] = ld_frag(&Wt[cur][cn][kg * 8], &Wt[cur][cn][16 + kg * 8]);
    }
#pragma unroll
    for (int t = 0; t < 4; ++t) acc[t] = wmma_bf16(af, bf[t], acc[t]);

    if (more) storeTile(cur ^ 1);
    cur ^= 1;
  }

  // epilogue: C layout M = r + 8*kg, N = lane&15
  float bv[4];
#pragma unroll
  for (int t = 0; t < 4; ++t)
    bv[t] = bias ? bias[n0 + wn * 64 + t * 16 + row] : 0.f;
#pragma unroll
  for (int r = 0; r < 8; ++r) {
    int gm = m0 + wm * 16 + r + 8 * kg;
    if (gm >= M) continue;
#pragma unroll
    for (int t = 0; t < 4; ++t) {
      float v0 = (acc[t][r] + bv[t]) * scale;
      if (relu) v0 = fmaxf(v0, 0.f);
      C[(size_t)gm * N + n0 + wn * 64 + t * 16 + row] = v0;
    }
  }
}

// ---------------------------------------------------------------------------
// Fused gated attention (flash style), DH = 64:
//   raw = (Q Kᵀ) * sigmoid(alphas) + (1 - sigmoid(alphas)) * ext
//   out = softmax_j(raw) @ V
// block = 256 = 8 waves; each wave owns one 16-row Q block; waves share
// double-buffered 32x64 K (row-major) and 64x32 Vᵀ chunks in LDS (bf16),
// one barrier per chunk.
// ---------------------------------------------------------------------------
__global__ __launch_bounds__(256) void flash_attn_kernel(
    const float* __restrict__ q, const float* __restrict__ k,
    const float* __restrict__ v, const float* __restrict__ alphas_l,
    const float* __restrict__ ext, float* __restrict__ out, int S, int H) {
  const int DH = 64;
  const int D  = H * DH;

  __shared__ __bf16 Ks[2][32][72];    // K chunk, row-major [j][d]
  __shared__ __bf16 Vt[2][64][40];    // V chunk, transposed [d][j]
  __shared__ __bf16 Pa[8][16][40];    // per-wave P chunk (A-layout staging)
  __shared__ float  Sc[8][16][36];    // per-wave raw scores (float4-aligned)
  __shared__ float  Alp[8][16];
  __shared__ float  Linv[8][16];

  const int tid   = threadIdx.x;
  const int lane  = tid & 31;
  const int wave  = tid >> 5;
  const int row   = lane & 15;
  const int kg    = lane >> 4;
  const int b     = blockIdx.z;
  const int h     = blockIdx.y;
  const int qbase = (blockIdx.x * 8 + wave) * 16;
  const bool active = qbase < S;

  // preload Q fragments (two K=32 chunks over DH=64) with float4 loads
  v16bf qa0, qa1;
  {
    int gm = qbase + row;
    if (gm >= S) gm = S - 1;
    const float* qp = q + ((size_t)b * S + gm) * D + h * DH;
#pragma unroll
    for (int ch = 0; ch < 2; ++ch) {
      const float* p = qp + ch * 32 + kg * 8;
      float4 a0 = *(const float4*)p;
      float4 a1 = *(const float4*)(p + 4);
      float4 c0 = *(const float4*)(p + 16);
      float4 c1 = *(const float4*)(p + 20);
      if (ch == 0) qa0 = pack16(a0, a1, c0, c1);
      else         qa1 = pack16(a0, a1, c0, c1);
    }
  }

  float m_i = -3.0e38f;  // row state lives in lanes 0..15 (row = lane)
  float l_i = 0.f;
  v8f acc[4] = {};

  // staging index precompute
  const int kr  = tid >> 3;            // K: j row 0..31
  const int kc0 = (tid & 7) * 8;       // K: d offset
  const int vd  = tid & 63;            // V: d 0..63
  const int vjq = (tid >> 6) * 8;      // V: j offset 0/8/16/24

  // chunk prefetch registers
  float4 pk0, pk1;
  float  pv[8];
  auto loadKV = [&](int jbase) {
    int gj = jbase + kr;
    bool ok = gj < S;
    const float* kp =
        k + ((size_t)b * S + (ok ? gj : (S - 1))) * D + h * DH + kc0;
    pk0 = *(const float4*)kp;
    pk1 = *(const float4*)(kp + 4);
    if (!ok) { pk0 = make_float4(0.f, 0.f, 0.f, 0.f); pk1 = pk0; }
    const float* vp = v + (size_t)b * S * D + h * DH + vd;
#pragma unroll
    for (int c = 0; c < 8; ++c) {
      int gjv = jbase + vjq + c;
      float val = vp[(size_t)(gjv < S ? gjv : (S - 1)) * D];
      pv[c] = (gjv < S) ? val : 0.f;
    }
  };
  auto storeKV = [&](int bufi) {
    *(v8bf*)&Ks[bufi][kr][kc0] = pack8(pk0, pk1);
    *(v8bf*)&Vt[bufi][vd][vjq] =
        pack8(make_float4(pv[0], pv[1], pv[2], pv[3]),
              make_float4(pv[4], pv[5], pv[6], pv[7]));
  };

  const int nchunks = (S + 31) / 32;
  loadKV(0);
  storeKV(0);
  int cur = 0;
  for (int jc = 0; jc < nchunks; ++jc) {
    const int jbase = jc * 32;
    __syncthreads();
    const bool more = (jc + 1) < nchunks;
    if (more) loadKV(jbase + 32);  // overlap next chunk's global latency

    if (active) {
      // scores 16x32: hoist the 4 B-fragments, then 4 WMMAs back-to-back
      v16bf kb[4];
#pragma unroll
      for (int ch = 0; ch < 2; ++ch) {
        int dk = ch * 32 + kg * 8;
        kb[ch * 2 + 0] = ld_frag(&Ks[cur][row][dk], &Ks[cur][row][dk + 16]);
        kb[ch * 2 + 1] =
            ld_frag(&Ks[cur][16 + row][dk], &Ks[cur][16 + row][dk + 16]);
      }
      v8f s0 = {};
      v8f s1 = {};
      s0 = wmma_bf16(qa0, kb[0], s0);
      s1 = wmma_bf16(qa0, kb[1], s1);
      s0 = wmma_bf16(qa1, kb[2], s0);
      s1 = wmma_bf16(qa1, kb[3], s1);
      // sigmoid gate + external matrix + tail mask -> per-wave LDS scratch
#pragma unroll
      for (int r = 0; r < 8; ++r) {
        int gm  = qbase + r + 8 * kg;
        int gj0 = jbase + row;
        int gj1 = jbase + 16 + row;
        float raw0 = -3.0e38f, raw1 = -3.0e38f;
        if (gm < S) {
          if (gj0 < S) {
            float a = alphas_l[(size_t)gm * S + gj0];
            float g = 1.f / (1.f + __expf(-a));
            float e = ext[((size_t)b * S + gm) * S + gj0];
            raw0 = s0[r] * g + (1.f - g) * e;
          }
          if (gj1 < S) {
            float a = alphas_l[(size_t)gm * S + gj1];
            float g = 1.f / (1.f + __expf(-a));
            float e = ext[((size_t)b * S + gm) * S + gj1];
            raw1 = s1[r] * g + (1.f - g) * e;
          }
        }
        Sc[wave][r + 8 * kg][row]      = raw0;
        Sc[wave][r + 8 * kg][16 + row] = raw1;
      }
      // online softmax: lane < 16 owns row = lane (same-wave LDS ordering)
      if (lane < 16) {
        float mloc = m_i;
#pragma unroll
        for (int c4 = 0; c4 < 8; ++c4) {
          float4 t = *(const float4*)&Sc[wave][lane][c4 * 4];
          mloc = fmaxf(mloc, fmaxf(fmaxf(t.x, t.y), fmaxf(t.z, t.w)));
        }
        float al   = __expf(m_i - mloc);
        float ssum = 0.f;
#pragma unroll
        for (int c8 = 0; c8 < 4; ++c8) {
          float4 t0 = *(const float4*)&Sc[wave][lane][c8 * 8];
          float4 t1 = *(const float4*)&Sc[wave][lane][c8 * 8 + 4];
          float4 p0, p1;
          p0.x = __expf(t0.x - mloc); p0.y = __expf(t0.y - mloc);
          p0.z = __expf(t0.z - mloc); p0.w = __expf(t0.w - mloc);
          p1.x = __expf(t1.x - mloc); p1.y = __expf(t1.y - mloc);
          p1.z = __expf(t1.z - mloc); p1.w = __expf(t1.w - mloc);
          ssum += p0.x + p0.y + p0.z + p0.w + p1.x + p1.y + p1.z + p1.w;
          *(v8bf*)&Pa[wave][lane][c8 * 8] = pack8(p0, p1);
        }
        l_i = l_i * al + ssum;
        m_i = mloc;
        Alp[wave][lane] = al;
      }
      // rescale running accumulators (vector reads of per-row alpha)
      float4 a0 = *(const float4*)&Alp[wave][kg * 8];
      float4 a1 = *(const float4*)&Alp[wave][kg * 8 + 4];
      float alf[8] = {a0.x, a0.y, a0.z, a0.w, a1.x, a1.y, a1.z, a1.w};
#pragma unroll
      for (int t = 0; t < 4; ++t)
#pragma unroll
        for (int r = 0; r < 8; ++r) acc[t][r] *= alf[r];
      // P(16x32) @ V(32x64): hoist P and V fragments, 4 WMMAs back-to-back
      v16bf pf = ld_frag(&Pa[wave][row][kg * 8], &Pa[wave][row][16 + kg * 8]);
      v16bf vf[4];
#pragma unroll
      for (int t = 0; t < 4; ++t) {
        const int vr = t * 16 + row;
        vf[t] = ld_frag(&Vt[cur][vr][kg * 8], &Vt[cur][vr][16 + kg * 8]);
      }
#pragma unroll
      for (int t = 0; t < 4; ++t) acc[t] = wmma_bf16(pf, vf[t], acc[t]);
    }
    if (more) storeKV(cur ^ 1);
    cur ^= 1;
  }

  if (active) {
    if (lane < 16) Linv[wave][lane] = (l_i > 0.f) ? (1.f / l_i) : 0.f;
    float4 l0 = *(const float4*)&Linv[wave][kg * 8];
    float4 l1 = *(const float4*)&Linv[wave][kg * 8 + 4];
    float li[8] = {l0.x, l0.y, l0.z, l0.w, l1.x, l1.y, l1.z, l1.w};
#pragma unroll
    for (int t = 0; t < 4; ++t)
#pragma unroll
      for (int r = 0; r < 8; ++r) {
        int gm = qbase + r + 8 * kg;
        if (gm < S)
          out[((size_t)b * S + gm) * D + h * DH + t * 16 + row] =
              acc[t][r] * li[r];
      }
  }
}

// ---------------------------------------------------------------------------
// x = LN(x + r) * g + b   (D = 512, one wave per row, float4 I/O)
// ---------------------------------------------------------------------------
__global__ __launch_bounds__(256) void residual_ln_kernel(
    float* __restrict__ x, const float* __restrict__ r,
    const float* __restrict__ g, const float* __restrict__ b, int Mrows) {
  const int D = 512;
  int wave = threadIdx.x >> 5, lane = threadIdx.x & 31;
  int rowi = blockIdx.x * 8 + wave;
  if (rowi >= Mrows) return;
  float* xp = x + (size_t)rowi * D + lane * 16;
  const float* rp = r + (size_t)rowi * D + lane * 16;
  float4 v[4];
  float s = 0.f;
#pragma unroll
  for (int c = 0; c < 4; ++c) {
    float4 xv = *(const float4*)(xp + 4 * c);
    float4 rv = *(const float4*)(rp + 4 * c);
    v[c] = make_float4(xv.x + rv.x, xv.y + rv.y, xv.z + rv.z, xv.w + rv.w);
    s += v[c].x + v[c].y + v[c].z + v[c].w;
  }
#pragma unroll
  for (int o = 16; o > 0; o >>= 1) s += __shfl_xor(s, o, 32);
  float mu = s * (1.f / D), var = 0.f;
#pragma unroll
  for (int c = 0; c < 4; ++c) {
    float dx = v[c].x - mu, dy = v[c].y - mu, dz = v[c].z - mu, dw = v[c].w - mu;
    var += dx * dx + dy * dy + dz * dz + dw * dw;
  }
#pragma unroll
  for (int o = 16; o > 0; o >>= 1) var += __shfl_xor(var, o, 32);
  float rs = rsqrtf(var * (1.f / D) + 1e-5f);
  const float* gp = g + lane * 16;
  const float* bp = b + lane * 16;
#pragma unroll
  for (int c = 0; c < 4; ++c) {
    float4 gv = *(const float4*)(gp + 4 * c);
    float4 bv = *(const float4*)(bp + 4 * c);
    float4 o;
    o.x = (v[c].x - mu) * rs * gv.x + bv.x;
    o.y = (v[c].y - mu) * rs * gv.y + bv.y;
    o.z = (v[c].z - mu) * rs * gv.z + bv.z;
    o.w = (v[c].w - mu) * rs * gv.w + bv.w;
    *(float4*)(xp + 4 * c) = o;
  }
}

// x = LN(LN(x + r, g1,b1), g2,b2)
__global__ __launch_bounds__(256) void residual_ln2_kernel(
    float* __restrict__ x, const float* __restrict__ r,
    const float* __restrict__ g1, const float* __restrict__ b1,
    const float* __restrict__ g2, const float* __restrict__ b2, int Mrows) {
  const int D = 512;
  int wave = threadIdx.x >> 5, lane = threadIdx.x & 31;
  int rowi = blockIdx.x * 8 + wave;
  if (rowi >= Mrows) return;
  float* xp = x + (size_t)rowi * D + lane * 16;
  const float* rp = r + (size_t)rowi * D + lane * 16;
  float4 v[4];
  float s = 0.f;
#pragma unroll
  for (int c = 0; c < 4; ++c) {
    float4 xv = *(const float4*)(xp + 4 * c);
    float4 rv = *(const float4*)(rp + 4 * c);
    v[c] = make_float4(xv.x + rv.x, xv.y + rv.y, xv.z + rv.z, xv.w + rv.w);
    s += v[c].x + v[c].y + v[c].z + v[c].w;
  }
#pragma unroll
  for (int o = 16; o > 0; o >>= 1) s += __shfl_xor(s, o, 32);
  float mu = s * (1.f / D), var = 0.f;
#pragma unroll
  for (int c = 0; c < 4; ++c) {
    float dx = v[c].x - mu, dy = v[c].y - mu, dz = v[c].z - mu, dw = v[c].w - mu;
    var += dx * dx + dy * dy + dz * dz + dw * dw;
  }
#pragma unroll
  for (int o = 16; o > 0; o >>= 1) var += __shfl_xor(var, o, 32);
  float rs = rsqrtf(var * (1.f / D) + 1e-5f);
  const float* g1p = g1 + lane * 16;
  const float* b1p = b1 + lane * 16;
#pragma unroll
  for (int c = 0; c < 4; ++c) {
    float4 gv = *(const float4*)(g1p + 4 * c);
    float4 bv = *(const float4*)(b1p + 4 * c);
    v[c].x = (v[c].x - mu) * rs * gv.x + bv.x;
    v[c].y = (v[c].y - mu) * rs * gv.y + bv.y;
    v[c].z = (v[c].z - mu) * rs * gv.z + bv.z;
    v[c].w = (v[c].w - mu) * rs * gv.w + bv.w;
  }
  s = 0.f;
#pragma unroll
  for (int c = 0; c < 4; ++c) s += v[c].x + v[c].y + v[c].z + v[c].w;
#pragma unroll
  for (int o = 16; o > 0; o >>= 1) s += __shfl_xor(s, o, 32);
  mu = s * (1.f / D); var = 0.f;
#pragma unroll
  for (int c = 0; c < 4; ++c) {
    float dx = v[c].x - mu, dy = v[c].y - mu, dz = v[c].z - mu, dw = v[c].w - mu;
    var += dx * dx + dy * dy + dz * dz + dw * dw;
  }
#pragma unroll
  for (int o = 16; o > 0; o >>= 1) var += __shfl_xor(var, o, 32);
  rs = rsqrtf(var * (1.f / D) + 1e-5f);
  const float* g2p = g2 + lane * 16;
  const float* b2p = b2 + lane * 16;
#pragma unroll
  for (int c = 0; c < 4; ++c) {
    float4 gv = *(const float4*)(g2p + 4 * c);
    float4 bv = *(const float4*)(b2p + 4 * c);
    float4 o;
    o.x = (v[c].x - mu) * rs * gv.x + bv.x;
    o.y = (v[c].y - mu) * rs * gv.y + bv.y;
    o.z = (v[c].z - mu) * rs * gv.z + bv.z;
    o.w = (v[c].w - mu) * rs * gv.w + bv.w;
    *(float4*)(xp + 4 * c) = o;
  }
}

// pooled[b,d] = mean_s x[b,s,d]
__global__ __launch_bounds__(256) void pool_kernel(const float* __restrict__ x,
                                                   float* __restrict__ pooled,
                                                   int S, int D) {
  int b = blockIdx.x;
  for (int d = threadIdx.x; d < D; d += blockDim.x) {
    float s = 0.f;
    for (int ss = 0; ss < S; ++ss) s += x[((size_t)b * S + ss) * D + d];
    pooled[(size_t)b * D + d] = s / (float)S;
  }
}

// ---------------------------------------------------------------------------
// Host orchestration
// ---------------------------------------------------------------------------
extern "C" void kernel_launch(void* const* d_in, const int* in_sizes, int n_in,
                              void* d_out, int out_size, void* d_ws,
                              size_t ws_size, hipStream_t stream) {
  (void)in_sizes; (void)n_in; (void)out_size; (void)ws_size;
  const int L = 4, D = 512, H = 8, FF = 1024, S = 933, DIN = 1280, DOUT = 768,
            Bb = 4;

  const float* pe     = (const float*)d_in[0];
  const float* ppi    = (const float*)d_in[1];
  const float* pfeat  = (const float*)d_in[2];
  const float* Wi     = (const float*)d_in[3];
  const float* bi     = (const float*)d_in[4];
  const float* qw     = (const float*)d_in[5];
  const float* qb     = (const float*)d_in[6];
  const float* kw     = (const float*)d_in[7];
  const float* kb     = (const float*)d_in[8];
  const float* vw     = (const float*)d_in[9];
  const float* vb     = (const float*)d_in[10];
  const float* ow     = (const float*)d_in[11];
  const float* ob     = (const float*)d_in[12];
  const float* w1     = (const float*)d_in[13];
  const float* b1     = (const float*)d_in[14];
  const float* w2     = (const float*)d_in[15];
  const float* b2     = (const float*)d_in[16];
  const float* ln1g   = (const float*)d_in[17];
  const float* ln1b   = (const float*)d_in[18];
  const float* ln2g   = (const float*)d_in[19];
  const float* ln2b   = (const float*)d_in[20];
  const float* lnog   = (const float*)d_in[21];
  const float* lnob   = (const float*)d_in[22];
  const float* alphas = (const float*)d_in[23];
  const float* Wo     = (const float*)d_in[24];
  const float* bo     = (const float*)d_in[25];
  float* outp = (float*)d_out;

  const size_t MS = (size_t)Bb * S;  // 3732 rows
  float* ws     = (float*)d_ws;
  float* x      = ws;                 // [MS, D]
  float* qB     = x + MS * D;         // [MS, D]
  float* kB     = qB + MS * D;        // [MS, D]
  float* vB     = kB + MS * D;        // [MS, D]
  float* tmp    = vB + MS * D;        // [MS, D]
  float* ffB    = tmp + MS * D;       // [MS, FF] (also holds attn output)
  float* pooled = ffB + MS * FF;      // [Bb, D]

  dim3 blk(256);
  auto gemm = [&](const float* A, const float* W, const float* bias, float* C,
                  int M, int N, int K, float scale, int relu) {
    dim3 grid((N + 127) / 128, (M + 63) / 64);
    gemm_bias_kernel<<<grid, blk, 0, stream>>>(A, W, bias, C, M, N, K, scale,
                                               relu);
  };

  // input projection
  gemm(pe, Wi, bi, x, (int)MS, D, DIN, 1.f, 0);

  const int rowblocks = (S + 15) / 16;
  dim3 agrid((rowblocks + 7) / 8, H, Bb);

  for (int i = 0; i < L; ++i) {
    const float* ext = (i % 2 == 0) ? pfeat : ppi;
    // Q (pre-scaled by 1/sqrt(DH)=1/8, bias included per reference), K, V
    gemm(x, qw + (size_t)i * D * D, qb + (size_t)i * D, qB, (int)MS, D, D,
         0.125f, 0);
    gemm(x, kw + (size_t)i * D * D, kb + (size_t)i * D, kB, (int)MS, D, D, 1.f,
         0);
    gemm(x, vw + (size_t)i * D * D, vb + (size_t)i * D, vB, (int)MS, D, D, 1.f,
         0);
    // fused gated attention -> ffB (first MS*D floats)
    flash_attn_kernel<<<agrid, blk, 0, stream>>>(
        qB, kB, vB, alphas + (size_t)i * S * S, ext, ffB, S, H);
    // output projection
    gemm(ffB, ow + (size_t)i * D * D, ob + (size_t)i * D, tmp, (int)MS, D, D,
         1.f, 0);
    // x = LN(x + attn)
    residual_ln_kernel<<<dim3(((int)MS + 7) / 8), blk, 0, stream>>>(
        x, tmp, ln1g + (size_t)i * D, ln1b + (size_t)i * D, (int)MS);
    // FF
    gemm(x, w1 + (size_t)i * D * FF, b1 + (size_t)i * FF, ffB, (int)MS, FF, D,
         1.f, 1);
    gemm(ffB, w2 + (size_t)i * FF * D, b2 + (size_t)i * D, tmp, (int)MS, D, FF,
         1.f, 0);
    // x = LN(LN(x + ff))
    residual_ln2_kernel<<<dim3(((int)MS + 7) / 8), blk, 0, stream>>>(
        x, tmp, ln2g + (size_t)i * D, ln2b + (size_t)i * D,
        lnog + (size_t)i * D, lnob + (size_t)i * D, (int)MS);
  }

  // pool over sequence, then final projection with ReLU into d_out
  pool_kernel<<<dim3(Bb), blk, 0, stream>>>(x, pooled, S, D);
  gemm(pooled, Wo, bo, outp, Bb, DOUT, D, 1.f, 1);
}
